// MultiHeadAttention_62139586839253
// MI455X (gfx1250) — matrix operand order repaired and checked by reading, not compiled, not run on hardware
//
#include <hip/hip_runtime.h>
#include <hip/hip_bf16.h>
#include <math.h>

#define EDIM 1024
#define NH   16
#define HD   64
#define BB   4
#define SS   2048
#define BS   (BB*SS)   // 8192

typedef __attribute__((ext_vector_type(16))) __bf16 bf16x16;
typedef __attribute__((ext_vector_type(8)))  __bf16 bf16x8;
typedef __attribute__((ext_vector_type(8)))  float  f32x8;
typedef __attribute__((ext_vector_type(4)))  int    i32x4;

union FragBF { bf16x16 v; bf16x8 h[2]; };

#if defined(__has_builtin)
#  if __has_builtin(__builtin_amdgcn_global_load_async_to_lds_b128)
#    define HAVE_ASYNC_LDS 1
#  endif
#  if __has_builtin(__builtin_amdgcn_s_wait_asynccnt)
#    define HAVE_WAIT_ASYNC 1
#  endif
#endif
#ifndef HAVE_ASYNC_LDS
#  define HAVE_ASYNC_LDS 0
#endif
#ifndef HAVE_WAIT_ASYNC
#  define HAVE_WAIT_ASYNC 0
#endif

// builtin signature (from compiler diagnostic): (int4 AS1*, int4 AS3*, imm, imm)
#define GQ(x) ((__attribute__((address_space(1))) i32x4*)(x))
#define LQ(x) ((__attribute__((address_space(3))) i32x4*)(x))

__device__ __forceinline__ void wait_async0() {
#if HAVE_ASYNC_LDS
# if HAVE_WAIT_ASYNC
  __builtin_amdgcn_s_wait_asynccnt(0);
# else
  asm volatile("s_wait_asynccnt 0" ::: "memory");
# endif
#endif
}

__device__ __forceinline__ f32x8 wmma_bf16(bf16x16 a, bf16x16 b, f32x8 c) {
  // 8 args: (neg_a, A, neg_b, B, c_mod, C, reuse_a, reuse_b)
  return __builtin_amdgcn_wmma_f32_16x16x32_bf16(false, a, false, b, (short)0, c, false, false);
}

// ---------------------------------------------------------------------------
// Kernel 1: P = X(f32, BSxE) @ W^T(f32, ExE), written as bf16 [B,H,S,D]
// 64(M) x 128(N) tile, 8 waves, K-step 32, double-buffered LDS (1 barrier/iter)
// ---------------------------------------------------------------------------
__global__ __launch_bounds__(256) void gemm_qkv_kernel(const float* __restrict__ X,
                                                       const float* __restrict__ W,
                                                       __bf16* __restrict__ Out) {
  __shared__ __bf16 Xs[2][64 * 40];
  __shared__ __bf16 Ws[2][128 * 40];

  const int tid  = threadIdx.x;
  const int lane = tid & 31;
  const int wave = tid >> 5;
  const int wm   = wave & 3;
  const int wn   = wave >> 2;
  const int lh   = (lane >= 16);
  const int l15  = lane & 15;
  const int mbase = blockIdx.x * 64;
  const int nbase = blockIdx.y * 128;

  f32x8 acc[4];
  #pragma unroll
  for (int i = 0; i < 4; i++) acc[i] = (f32x8){0,0,0,0,0,0,0,0};

  const int xr = tid >> 2, xc = (tid & 3) * 8;
  const int wr = tid >> 1, wc = (tid & 1) * 16;
  const float* Xbase = X + (size_t)(mbase + xr) * EDIM + xc;
  const float* Wbase = W + (size_t)(nbase + wr) * EDIM + wc;

  float fx[8], fw[16];
  auto load_glb = [&](int kk) {
    #pragma unroll
    for (int e = 0; e < 8;  e++) fx[e] = Xbase[kk + e];
    #pragma unroll
    for (int e = 0; e < 16; e++) fw[e] = Wbase[kk + e];
  };
  auto store_lds = [&](int p) {
    bf16x8 t;
    #pragma unroll
    for (int e = 0; e < 8; e++) t[e] = (__bf16)fx[e];
    *(bf16x8*)&Xs[p][xr * 40 + xc] = t;
    bf16x8 t0, t1;
    #pragma unroll
    for (int e = 0; e < 8; e++) { t0[e] = (__bf16)fw[e]; t1[e] = (__bf16)fw[8 + e]; }
    *(bf16x8*)&Ws[p][wr * 40 + wc]     = t0;
    *(bf16x8*)&Ws[p][wr * 40 + wc + 8] = t1;
  };

  load_glb(0);
  store_lds(0);
  __syncthreads();

  const int am = 16 * wm + l15;
  const int ab = lh ? 16 : 0;
  const int bb = lh ? 16 : 0;

  int p = 0;
  for (int kk = 0; kk < EDIM; kk += 32) {
    const bool more = (kk + 32 < EDIM);
    if (more) load_glb(kk + 32);          // overlap HBM latency with WMMA

    FragBF A;
    A.h[0] = *(const bf16x8*)&Xs[p][am * 40 + ab];
    A.h[1] = *(const bf16x8*)&Xs[p][am * 40 + ab + 8];
    FragBF Bf[4];
    #pragma unroll
    for (int nf = 0; nf < 4; nf++) {
      const int bn = 64 * wn + 16 * nf + l15;
      Bf[nf].h[0] = *(const bf16x8*)&Ws[p][bn * 40 + bb];
      Bf[nf].h[1] = *(const bf16x8*)&Ws[p][bn * 40 + bb + 8];
    }
    #pragma unroll
    for (int nf = 0; nf < 4; nf++) acc[nf] = wmma_bf16(A.v, Bf[nf].v, acc[nf]);

    if (more) store_lds(p ^ 1);
    __syncthreads();
    p ^= 1;
  }

  // Epilogue: D layout -> bf16 [B, H, S, D]
  #pragma unroll
  for (int nf = 0; nf < 4; nf++) {
    #pragma unroll
    for (int r = 0; r < 8; r++) {
      const int m = mbase + 16 * wm + r + (lh ? 8 : 0);
      const int n = nbase + 64 * wn + 16 * nf + l15;
      const int b = m >> 11, s = m & 2047;
      const int h = n >> 6,  d = n & 63;
      Out[(((size_t)b * NH + h) * SS + s) * HD + d] = (__bf16)acc[nf][r];
    }
  }
}

// ---------------------------------------------------------------------------
// Kernel 2: in-place RoPE on Q/K, bf16 [B,H,S,D]; thread per (even,odd) pair
// ---------------------------------------------------------------------------
__global__ __launch_bounds__(256) void rope_kernel(__bf16* __restrict__ Qh,
                                                   __bf16* __restrict__ Kh) {
  const int idx = blockIdx.x * 256 + threadIdx.x;
  __bf16* P = (blockIdx.y == 0) ? Qh : Kh;
  const int d2 = idx & 31;
  const int s  = (idx >> 5) & 2047;
  const int bh = idx >> 16;
  const size_t base = ((size_t)bh * SS + s) * HD + d2 * 2;
  const float inv_freq = __expf(-(float)(2 * d2) * (9.2103403719761836f / 64.0f));
  const float ang = (float)s * inv_freq;
  float sn, c;
  __sincosf(ang, &sn, &c);
  const float x0 = (float)P[base], x1 = (float)P[base + 1];
  P[base]     = (__bf16)(x0 * c  - x1 * sn);
  P[base + 1] = (__bf16)(x0 * sn + x1 * c);
}

// ---------------------------------------------------------------------------
// Kernel 3: flash attention per (b, h, 64 q-rows). 4 waves, 16 q-rows/wave.
// 32-key chunks; K staged via async global->LDS copies (ASYNCcnt), V staged
// transposed through VGPRs; double-buffered; online softmax; 8 WMMAs/chunk.
// ---------------------------------------------------------------------------
__global__ __launch_bounds__(128) void attn_kernel(const __bf16* __restrict__ Qh,
                                                   const __bf16* __restrict__ Kh,
                                                   const __bf16* __restrict__ Vh,
                                                   __bf16* __restrict__ Oh) {
  __shared__ __bf16 Ks[2][32 * 72];
  __shared__ __bf16 Vt[2][64 * 40];
  __shared__ __bf16 Pb[4][16 * 40];

  const int tid  = threadIdx.x;
  const int lane = tid & 31;
  const int wave = tid >> 5;
  const int lh   = (lane >= 16);
  const int l15  = lane & 15;
  const int b = blockIdx.z, h = blockIdx.y;
  const size_t bh = ((size_t)b * NH + h) * SS * HD;
  const int q0 = blockIdx.x * 64 + wave * 16;
  const float SC = 0.125f;

  const int jj = tid >> 2, pcol = (tid & 3) * 16;   // staging split: 4 thr/key
  const __bf16* Kc = Kh + bh + (size_t)jj * HD + pcol;
  const __bf16* Vc = Vh + bh + (size_t)jj * HD + pcol;

  bf16x8 vr0, vr1;  // V staging registers
  auto issue_K = [&](int j0, int p) {
#if HAVE_ASYNC_LDS
    __builtin_amdgcn_global_load_async_to_lds_b128(
        GQ(Kc + (size_t)j0 * HD), LQ(&Ks[p][jj * 72 + pcol]), 0, 0);
    __builtin_amdgcn_global_load_async_to_lds_b128(
        GQ(Kc + (size_t)j0 * HD + 8), LQ(&Ks[p][jj * 72 + pcol + 8]), 0, 0);
#else
    *(bf16x8*)&Ks[p][jj * 72 + pcol]     = *(const bf16x8*)(Kc + (size_t)j0 * HD);
    *(bf16x8*)&Ks[p][jj * 72 + pcol + 8] = *(const bf16x8*)(Kc + (size_t)j0 * HD + 8);
#endif
  };
  auto load_V = [&](int j0) {
    vr0 = *(const bf16x8*)(Vc + (size_t)j0 * HD);
    vr1 = *(const bf16x8*)(Vc + (size_t)j0 * HD + 8);
  };
  auto store_V = [&](int p) {
    #pragma unroll
    for (int e = 0; e < 8; e++) {
      Vt[p][(pcol + e) * 40 + jj]     = vr0[e];
      Vt[p][(pcol + 8 + e) * 40 + jj] = vr1[e];
    }
  };

  // Q A-fragments for d=[0,32) and d=[32,64)
  FragBF a0, a1;
  {
    const __bf16* qrow = Qh + bh + (size_t)(q0 + l15) * HD;
    const int ab = lh ? 16 : 0;
    a0.h[0] = *(const bf16x8*)&qrow[ab];
    a0.h[1] = *(const bf16x8*)&qrow[ab + 8];
    a1.h[0] = *(const bf16x8*)&qrow[32 + ab];
    a1.h[1] = *(const bf16x8*)&qrow[32 + ab + 8];
  }

  float rowM[8], rowL[8];
  #pragma unroll
  for (int r = 0; r < 8; r++) { rowM[r] = -INFINITY; rowL[r] = 0.0f; }
  f32x8 o[4];
  #pragma unroll
  for (int i = 0; i < 4; i++) o[i] = (f32x8){0,0,0,0,0,0,0,0};

  issue_K(0, 0);
  load_V(0);
  store_V(0);
  wait_async0();
  __syncthreads();

  int p = 0;
  for (int j0 = 0; j0 < SS; j0 += 32) {
    const bool more = (j0 + 32 < SS);
    if (more) { issue_K(j0 + 32, p ^ 1); load_V(j0 + 32); }

    // scores: two 16x16 tiles, K-dim = d (2 chunks of 32)
    f32x8 s0 = (f32x8){0,0,0,0,0,0,0,0};
    f32x8 s1 = (f32x8){0,0,0,0,0,0,0,0};
    #pragma unroll
    for (int kc = 0; kc < 2; kc++) {
      FragBF B0, B1;
      const int off = kc * 32 + (lh ? 16 : 0);
      B0.h[0] = *(const bf16x8*)&Ks[p][(l15)      * 72 + off];
      B0.h[1] = *(const bf16x8*)&Ks[p][(l15)      * 72 + off + 8];
      B1.h[0] = *(const bf16x8*)&Ks[p][(16 + l15) * 72 + off];
      B1.h[1] = *(const bf16x8*)&Ks[p][(16 + l15) * 72 + off + 8];
      const bf16x16 av = kc ? a1.v : a0.v;
      s0 = wmma_bf16(av, B0.v, s0);
      s1 = wmma_bf16(av, B1.v, s1);
    }

    // online softmax; rows m = r + 8*lh across the 16 lanes of a half
    float alpha[8];
    #pragma unroll
    for (int r = 0; r < 8; r++) {
      float v = fmaxf(s0[r], s1[r]) * SC;
      #pragma unroll
      for (int msk = 1; msk < 16; msk <<= 1) v = fmaxf(v, __shfl_xor(v, msk, 32));
      const float mn = fmaxf(rowM[r], v);
      const float al = __expf(rowM[r] - mn);
      const float p0 = __expf(s0[r] * SC - mn);
      const float p1 = __expf(s1[r] * SC - mn);
      float rs = p0 + p1;
      #pragma unroll
      for (int msk = 1; msk < 16; msk <<= 1) rs += __shfl_xor(rs, msk, 32);
      rowL[r] = rowL[r] * al + rs;
      rowM[r] = mn;
      alpha[r] = al;
      const int pm = r + (lh ? 8 : 0);
      Pb[wave][pm * 40 + l15]      = (__bf16)p0;
      Pb[wave][pm * 40 + 16 + l15] = (__bf16)p1;
    }
    #pragma unroll
    for (int dsb = 0; dsb < 4; dsb++)
      #pragma unroll
      for (int r = 0; r < 8; r++) o[dsb][r] *= alpha[r];

    // wave-local LDS RAW: DS in-order per wave; block compiler reordering
    asm volatile("s_wait_dscnt 0" ::: "memory");

    FragBF PA;
    const int ab = lh ? 16 : 0;
    PA.h[0] = *(const bf16x8*)&Pb[wave][l15 * 40 + ab];
    PA.h[1] = *(const bf16x8*)&Pb[wave][l15 * 40 + ab + 8];
    FragBF BV[4];
    #pragma unroll
    for (int dsb = 0; dsb < 4; dsb++) {
      const int off = lh ? 16 : 0;
      BV[dsb].h[0] = *(const bf16x8*)&Vt[p][(16 * dsb + l15) * 40 + off];
      BV[dsb].h[1] = *(const bf16x8*)&Vt[p][(16 * dsb + l15) * 40 + off + 8];
    }
    #pragma unroll
    for (int dsb = 0; dsb < 4; dsb++) o[dsb] = wmma_bf16(PA.v, BV[dsb].v, o[dsb]);

    if (more) store_V(p ^ 1);
    wait_async0();
    __syncthreads();
    p ^= 1;
  }

  #pragma unroll
  for (int r = 0; r < 8; r++) rowL[r] = 1.0f / rowL[r];
  #pragma unroll
  for (int dsb = 0; dsb < 4; dsb++) {
    #pragma unroll
    for (int r = 0; r < 8; r++) {
      const int s   = q0 + r + (lh ? 8 : 0);
      const int col = h * HD + 16 * dsb + l15;
      Oh[((size_t)b * SS + s) * EDIM + col] = (__bf16)(o[dsb][r] * rowL[r]);
    }
  }
}

// ---------------------------------------------------------------------------
// Kernel 4: Y(f32) = A(bf16, BSxE) @ Wo^T + bo; X staged via async->LDS
// ---------------------------------------------------------------------------
__global__ __launch_bounds__(256) void gemm_out_kernel(const __bf16* __restrict__ A,
                                                       const float* __restrict__ W,
                                                       const float* __restrict__ bias,
                                                       float* __restrict__ Y) {
  __shared__ __bf16 Xs[2][64 * 40];
  __shared__ __bf16 Ws[2][128 * 40];

  const int tid  = threadIdx.x;
  const int lane = tid & 31;
  const int wave = tid >> 5;
  const int wm   = wave & 3;
  const int wn   = wave >> 2;
  const int lh   = (lane >= 16);
  const int l15  = lane & 15;
  const int mbase = blockIdx.x * 64;
  const int nbase = blockIdx.y * 128;

  f32x8 acc[4];
  #pragma unroll
  for (int i = 0; i < 4; i++) acc[i] = (f32x8){0,0,0,0,0,0,0,0};

  const int xr = tid >> 2, xc = (tid & 3) * 8;
  const int wr = tid >> 1, wc = (tid & 1) * 16;
  const __bf16* Abase = A + (size_t)(mbase + xr) * EDIM + xc;
  const float*  Wbase = W + (size_t)(nbase + wr) * EDIM + wc;

  float fw[16];
  auto issue_X = [&](int kk, int p) {
#if HAVE_ASYNC_LDS
    __builtin_amdgcn_global_load_async_to_lds_b128(
        GQ(Abase + kk), LQ(&Xs[p][xr * 40 + xc]), 0, 0);
#else
    *(bf16x8*)&Xs[p][xr * 40 + xc] = *(const bf16x8*)(Abase + kk);
#endif
  };
  auto load_W = [&](int kk) {
    #pragma unroll
    for (int e = 0; e < 16; e++) fw[e] = Wbase[kk + e];
  };
  auto store_W = [&](int p) {
    bf16x8 t0, t1;
    #pragma unroll
    for (int e = 0; e < 8; e++) { t0[e] = (__bf16)fw[e]; t1[e] = (__bf16)fw[8 + e]; }
    *(bf16x8*)&Ws[p][wr * 40 + wc]     = t0;
    *(bf16x8*)&Ws[p][wr * 40 + wc + 8] = t1;
  };

  issue_X(0, 0);
  load_W(0);
  store_W(0);
  wait_async0();
  __syncthreads();

  const int am = 16 * wm + l15;
  const int ab = lh ? 16 : 0;
  const int bb = lh ? 16 : 0;

  int p = 0;
  for (int kk = 0; kk < EDIM; kk += 32) {
    const bool more = (kk + 32 < EDIM);
    if (more) { issue_X(kk + 32, p ^ 1); load_W(kk + 32); }

    FragBF Af;
    Af.h[0] = *(const bf16x8*)&Xs[p][am * 40 + ab];
    Af.h[1] = *(const bf16x8*)&Xs[p][am * 40 + ab + 8];
    FragBF Bf[4];
    #pragma unroll
    for (int nf = 0; nf < 4; nf++) {
      const int bn = 64 * wn + 16 * nf + l15;
      Bf[nf].h[0] = *(const bf16x8*)&Ws[p][bn * 40 + bb];
      Bf[nf].h[1] = *(const bf16x8*)&Ws[p][bn * 40 + bb + 8];
    }
    #pragma unroll
    for (int nf = 0; nf < 4; nf++) acc[nf] = wmma_bf16(Af.v, Bf[nf].v, acc[nf]);

    if (more) store_W(p ^ 1);
    wait_async0();
    __syncthreads();
    p ^= 1;
  }

  #pragma unroll
  for (int nf = 0; nf < 4; nf++) {
    const int n = nbase + 64 * wn + 16 * nf + l15;
    const float bv = bias[n];
    #pragma unroll
    for (int r = 0; r < 8; r++) {
      const int m = mbase + 16 * wm + r + (lh ? 8 : 0);
      Y[(size_t)m * EDIM + n] = acc[nf][r] + bv;
    }
  }
}

// ---------------------------------------------------------------------------
extern "C" void kernel_launch(void* const* d_in, const int* in_sizes, int n_in,
                              void* d_out, int out_size, void* d_ws, size_t ws_size,
                              hipStream_t stream) {
  (void)in_sizes; (void)n_in; (void)out_size; (void)ws_size;
  const float* q  = (const float*)d_in[0];
  const float* k  = (const float*)d_in[1];
  const float* v  = (const float*)d_in[2];
  const float* Wq = (const float*)d_in[3];
  const float* Wk = (const float*)d_in[4];
  const float* Wv = (const float*)d_in[5];
  const float* Wo = (const float*)d_in[6];
  const float* bo = (const float*)d_in[7];
  float* out = (float*)d_out;

  const size_t tensor_elems = (size_t)BB * NH * SS * HD;   // 8Mi elems
  __bf16* Qh = (__bf16*)d_ws;
  __bf16* Kh = Qh + tensor_elems;
  __bf16* Vh = Kh + tensor_elems;
  __bf16* Ao = Vh + tensor_elems;                          // 64MiB total

  dim3 gg(BS / 64, EDIM / 128);
  gemm_qkv_kernel<<<gg, 256, 0, stream>>>(q, Wq, Qh);
  gemm_qkv_kernel<<<gg, 256, 0, stream>>>(k, Wk, Kh);
  gemm_qkv_kernel<<<gg, 256, 0, stream>>>(v, Wv, Vh);

  const int pairs = BB * NH * SS * (HD / 2);               // 4,194,304
  rope_kernel<<<dim3(pairs / 256, 2), 256, 0, stream>>>(Qh, Kh);

  attn_kernel<<<dim3(SS / 64, NH, BB), 128, 0, stream>>>(Qh, Kh, Vh, Ao);

  gemm_out_kernel<<<gg, 256, 0, stream>>>(Ao, Wo, bo, out);
}
